// Router_58858231824505
// MI455X (gfx1250) — matrix operand order repaired and checked by reading, compiled, not run on hardware
//
#include <hip/hip_runtime.h>
#include <hip/hip_bf16.h>
#include <stdint.h>

// ---------------------------------------------------------------------------
// MI455X (gfx1250, wave32) Router kernel.
//   phase 0: transpose+convert weights to bf16, N-major (B-operand friendly)
//   phase 1: selector MLP via v_wmma_f32_16x16x32_bf16 + LayerNorm stats +
//            gumbel hard selection; writes A = s * (xn*gamma_e + beta_e) bf16
//   phase 2: head GEMM via WMMA, per-expert lane masks, bias epilogue
// ---------------------------------------------------------------------------

typedef __attribute__((ext_vector_type(16))) __bf16 v16bf;
typedef __attribute__((ext_vector_type(8)))  __bf16 bf16x8;
typedef __attribute__((ext_vector_type(8)))  float  f32x8;

union ABFrag {
  v16bf  v;
  bf16x8 h[2];
  uint4  u[2];
};

__device__ __forceinline__ unsigned short f2bf(float f) {
  union { float f; uint32_t u; } c; c.f = f;
  uint32_t u = c.u;
  u += 0x7FFFu + ((u >> 16) & 1u);   // round-to-nearest-even
  return (unsigned short)(u >> 16);
}

__device__ __forceinline__ bool wave_any(int pred) {
#if __has_builtin(__builtin_amdgcn_ballot_w32)
  return __builtin_amdgcn_ballot_w32(pred != 0) != 0u;
#else
  return __any(pred);
#endif
}

static constexpr int kB  = 32768;
static constexpr int kD  = 768;
static constexpr int kH  = 128;
static constexpr int kC  = 474;
static constexpr int kCP = 512;   // head_wT padded to 512 columns
static constexpr int kE  = 3;

// ---------------- phase 0: weight conversion ------------------------------

// sel_w1 [D][H] f32  ->  w1t [H][D] bf16   (N-major, B-operand layout)
__global__ __launch_bounds__(256) void k_prep_w1t(const float* __restrict__ w1,
                                                  unsigned short* __restrict__ w1t) {
  int i = blockIdx.x * 256 + threadIdx.x;
  if (i >= kH * kD) return;
  int d = i % kD, h = i / kD;
  w1t[i] = f2bf(w1[d * kH + h]);
}

// head_w [E][D][C] f32 -> wt [E][CP][D] bf16 (padded columns zero-filled)
__global__ __launch_bounds__(256) void k_prep_wt(const float* __restrict__ hw,
                                                 unsigned short* __restrict__ wt) {
  int i = blockIdx.x * 256 + threadIdx.x;
  if (i >= kE * kCP * kD) return;
  int d = i % kD;
  int c = (i / kD) % kCP;
  int e = i / (kD * kCP);
  float v = (c < kC) ? hw[((size_t)e * kD + d) * kC + c] : 0.0f;
  wt[i] = f2bf(v);
}

// ---------------- phase 1: selector + LayerNorm + selection ---------------
// 128 threads (4 waves) per block, 16 rows per block.

__global__ __launch_bounds__(128)
void k_selector(const float* __restrict__ x, const float* __restrict__ gn,
                const unsigned short* __restrict__ w1t,
                const float* __restrict__ b1, const float* __restrict__ w2,
                const float* __restrict__ b2, const float* __restrict__ gamma,
                const float* __restrict__ beta,
                unsigned short* __restrict__ xs_out,
                int* __restrict__ sel_out, float* __restrict__ s_out) {
  constexpr int XPITCH = kD + 8;          // pad LDS row to dodge bank conflicts
  __shared__ __align__(16) unsigned short xbf[16][XPITCH];
  __shared__ float hsh[16][kH];
  __shared__ float muS[16], rsS[16], sS[16];
  __shared__ int   selS[16];

  const int tid = threadIdx.x;
  const int rowBase = blockIdx.x * 16;
  const int r  = tid >> 3;                // 8 threads per row
  const int l8 = tid & 7;
  const float* xr = x + (size_t)(rowBase + r) * kD;

  // ---- pass 1: load x, stage bf16(x) in LDS, row sums for LayerNorm ----
  float sum = 0.f, ss = 0.f;
  #pragma unroll 4
  for (int i = 0; i < 24; ++i) {
    int c = i * 32 + l8 * 4;
    float4 v = *(const float4*)(xr + c);
    sum += v.x + v.y + v.z + v.w;
    ss  += v.x * v.x + v.y * v.y + v.z * v.z + v.w * v.w;
    xbf[r][c + 0] = f2bf(v.x); xbf[r][c + 1] = f2bf(v.y);
    xbf[r][c + 2] = f2bf(v.z); xbf[r][c + 3] = f2bf(v.w);
  }
  sum += __shfl_xor(sum, 1); ss += __shfl_xor(ss, 1);
  sum += __shfl_xor(sum, 2); ss += __shfl_xor(ss, 2);
  sum += __shfl_xor(sum, 4); ss += __shfl_xor(ss, 4);
  if (l8 == 0) {
    float mu  = sum * (1.0f / kD);
    float var = ss * (1.0f / kD) - mu * mu;
    muS[r] = mu;
    rsS[r] = rsqrtf(var + 1e-5f);
  }
  __syncthreads();

  // ---- selector GEMM: [16 x 768] x [768 x 128], 4 waves x 2 N-tiles ----
  const int wave = tid >> 5, lane = tid & 31;
  const int mrow  = lane & 15;
  const int koff  = (lane >> 4) * 16;        // A: K-half per half-wave
  const int bkoff = (lane < 16) ? 0 : 16;    // B: K-half per half-wave
  const int c0 = wave * 32 + (lane & 15);
  f32x8 acc0 = {}, acc1 = {};
  for (int kb = 0; kb < 24; ++kb) {
    int k = kb * 32;
    ABFrag a, bA, bB;
    a.h[0] = *(const bf16x8*)&xbf[mrow][k + koff];
    a.h[1] = *(const bf16x8*)&xbf[mrow][k + koff + 8];
    const unsigned short* p0 = w1t + (size_t)c0 * kD + k + bkoff;
    const unsigned short* p1 = w1t + (size_t)(c0 + 16) * kD + k + bkoff;
    bA.h[0] = *(const bf16x8*)(p0);
    bA.h[1] = *(const bf16x8*)(p0 + 8);
    bB.h[0] = *(const bf16x8*)(p1);
    bB.h[1] = *(const bf16x8*)(p1 + 8);
    acc0 = __builtin_amdgcn_wmma_f32_16x16x32_bf16(false, a.v, false, bA.v,
                                                   (short)0, acc0, false, false);
    acc1 = __builtin_amdgcn_wmma_f32_16x16x32_bf16(false, a.v, false, bB.v,
                                                   (short)0, acc1, false, false);
  }
  // C layout: VGPR i -> row i (+8 for lanes 16..31), col = lane%16..
  {
    int crow = (lane >> 4) * 8;
    int ccol = lane & 15;
    #pragma unroll
    for (int i = 0; i < 8; ++i) {
      int g0 = wave * 32 + ccol, g1 = g0 + 16;
      hsh[crow + i][g0] = fmaxf(acc0[i] + b1[g0], 0.0f);
      hsh[crow + i][g1] = fmaxf(acc1[i] + b1[g1], 0.0f);
    }
  }
  __syncthreads();

  // ---- second layer (128x3) + gumbel hard selection, 16 lanes ----
  if (tid < 16) {
    int rg = rowBase + tid;
    float l0 = b2[0], l1 = b2[1], l2 = b2[2];
    for (int k = 0; k < kH; ++k) {
      float hv = hsh[tid][k];
      l0 = fmaf(hv, w2[k * 3 + 0], l0);
      l1 = fmaf(hv, w2[k * 3 + 1], l1);
      l2 = fmaf(hv, w2[k * 3 + 2], l2);
    }
    l0 += gn[rg * 3 + 0]; l1 += gn[rg * 3 + 1]; l2 += gn[rg * 3 + 2];
    float m = fmaxf(l0, fmaxf(l1, l2));
    float e0 = __expf(l0 - m), e1 = __expf(l1 - m), e2 = __expf(l2 - m);
    float inv = 1.0f / (e0 + e1 + e2);
    float y0 = e0 * inv, y1 = e1 * inv, y2 = e2 * inv;
    int e = 0; float best = y0;
    if (y1 > best) { best = y1; e = 1; }
    if (y2 > best) { best = y2; e = 2; }
    float s = (1.0f - best) + best;        // straight-through forward value
    selS[tid] = e; sS[tid] = s;
    sel_out[rg] = e; s_out[rg] = s;
  }
  __syncthreads();

  // ---- pass 2: write A = s * (xn * gamma_e + beta_e) as bf16 ----
  {
    float mu = muS[r], rs = rsS[r], s = sS[r];
    const float* g  = gamma + (size_t)selS[r] * kD;
    const float* bb = beta  + (size_t)selS[r] * kD;
    unsigned short* xso = xs_out + (size_t)(rowBase + r) * kD;
    #pragma unroll 4
    for (int i = 0; i < 24; ++i) {
      int c = i * 32 + l8 * 4;
      float4 v  = *(const float4*)(xr + c);   // L2-hot reload
      float4 gv = *(const float4*)(g + c);
      float4 bv = *(const float4*)(bb + c);
      ushort4 o;
      o.x = f2bf(s * fmaf((v.x - mu) * rs, gv.x, bv.x));
      o.y = f2bf(s * fmaf((v.y - mu) * rs, gv.y, bv.y));
      o.z = f2bf(s * fmaf((v.z - mu) * rs, gv.z, bv.z));
      o.w = f2bf(s * fmaf((v.w - mu) * rs, gv.w, bv.w));
      *(ushort4*)(xso + c) = o;
    }
  }
}

// ---------------- phase 2: head GEMM ------------------------------------
// 256 threads = 8 waves tiled 4(M) x 2(N); each wave: 16 rows x 32 cols.

__global__ __launch_bounds__(256)
void k_heads(const unsigned short* __restrict__ xs,
             const unsigned short* __restrict__ wt,     // [E][kCP][kD] bf16
             const float* __restrict__ head_b,          // [E][kC] f32
             const int* __restrict__ sel, const float* __restrict__ sc,
             float* __restrict__ out) {
  const int tid = threadIdx.x;
  const int wave = tid >> 5, lane = tid & 31;
  const int wm = wave >> 1, wn = wave & 1;
  const int rowBase = blockIdx.y * 64 + wm * 16;
  const int colBase = blockIdx.x * 64 + wn * 32;

  const int mrow  = lane & 15;
  const int koff  = (lane >> 4) * 16;
  const int bkoff = (lane < 16) ? 0 : 16;
  const int c0 = colBase + (lane & 15);
  const int arow = rowBase + mrow;
  const int mySel = sel[arow];
  const unsigned short* xrow = xs + (size_t)arow * kD;

  f32x8 acc0 = {}, acc1 = {};
  for (int e = 0; e < kE; ++e) {
    if (!wave_any(mySel == e)) continue;        // wave-uniform skip
    const uint32_t msk = (mySel == e) ? 0xFFFFFFFFu : 0u;
    const unsigned short* w0 = wt + ((size_t)e * kCP + c0) * kD;
    const unsigned short* w1 = w0 + (size_t)16 * kD;
    for (int kb = 0; kb < 24; ++kb) {
      int k = kb * 32;
      ABFrag a, bA, bB;
      a.u[0] = *(const uint4*)(xrow + k + koff);
      a.u[1] = *(const uint4*)(xrow + k + koff + 8);
      a.u[0].x &= msk; a.u[0].y &= msk; a.u[0].z &= msk; a.u[0].w &= msk;
      a.u[1].x &= msk; a.u[1].y &= msk; a.u[1].z &= msk; a.u[1].w &= msk;
      bA.h[0] = *(const bf16x8*)(w0 + k + bkoff);
      bA.h[1] = *(const bf16x8*)(w0 + k + bkoff + 8);
      bB.h[0] = *(const bf16x8*)(w1 + k + bkoff);
      bB.h[1] = *(const bf16x8*)(w1 + k + bkoff + 8);
      __builtin_prefetch(w0 + k + 128, 0, 1);   // global_prefetch next tiles
      __builtin_prefetch(w1 + k + 128, 0, 1);
      acc0 = __builtin_amdgcn_wmma_f32_16x16x32_bf16(false, a.v, false, bA.v,
                                                     (short)0, acc0, false, false);
      acc1 = __builtin_amdgcn_wmma_f32_16x16x32_bf16(false, a.v, false, bB.v,
                                                     (short)0, acc1, false, false);
    }
  }

  // epilogue: + s_b * head_b[e_b], bounds-checked store (C = 474)
  const int crow = (lane >> 4) * 8;
  const int ccol = lane & 15;
  #pragma unroll
  for (int i = 0; i < 8; ++i) {
    int rr = rowBase + crow + i;
    int e  = sel[rr];
    float s = sc[rr];
    int cg0 = colBase + ccol;
    int cg1 = cg0 + 16;
    if (cg0 < kC) out[(size_t)rr * kC + cg0] = acc0[i] + s * head_b[(size_t)e * kC + cg0];
    if (cg1 < kC) out[(size_t)rr * kC + cg1] = acc1[i] + s * head_b[(size_t)e * kC + cg1];
  }
}

// ---------------- launch ---------------------------------------------------

extern "C" void kernel_launch(void* const* d_in, const int* in_sizes, int n_in,
                              void* d_out, int out_size, void* d_ws, size_t ws_size,
                              hipStream_t stream) {
  const float* x     = (const float*)d_in[0];
  const float* gn    = (const float*)d_in[1];
  const float* w1    = (const float*)d_in[2];
  const float* b1    = (const float*)d_in[3];
  const float* w2    = (const float*)d_in[4];
  const float* b2    = (const float*)d_in[5];
  const float* gamma = (const float*)d_in[6];
  const float* beta  = (const float*)d_in[7];
  const float* hw    = (const float*)d_in[8];
  const float* hb    = (const float*)d_in[9];
  float* out = (float*)d_out;

  // workspace layout (~53 MB): xs bf16 | w1t bf16 | head_wT bf16 | sel | scale
  char* ws = (char*)d_ws;
  size_t off = 0;
  auto take = [&](size_t bytes) {
    char* p = ws + off;
    off = (off + bytes + 255) & ~(size_t)255;
    return p;
  };
  unsigned short* xs  = (unsigned short*)take((size_t)kB * kD * 2);
  unsigned short* w1t = (unsigned short*)take((size_t)kH * kD * 2);
  unsigned short* wt  = (unsigned short*)take((size_t)kE * kCP * kD * 2);
  int*   sel = (int*)take((size_t)kB * 4);
  float* sc  = (float*)take((size_t)kB * 4);

  k_prep_w1t<<<(kH * kD + 255) / 256, 256, 0, stream>>>(w1, w1t);
  k_prep_wt<<<(kE * kCP * kD + 255) / 256, 256, 0, stream>>>(hw, wt);
  k_selector<<<kB / 16, 128, 0, stream>>>(x, gn, w1t, b1, w2, b2, gamma, beta,
                                          xs, sel, sc);
  dim3 grid((kC + 63) / 64, kB / 64);
  k_heads<<<grid, 256, 0, stream>>>(xs, wt, hb, sel, sc, out);
}
